// Model_2869038154194
// MI455X (gfx1250) — matrix operand order repaired and checked by reading, not compiled, not run on hardware
//
#include <hip/hip_runtime.h>
#include <math.h>

typedef _Float16 half_t;
typedef __attribute__((ext_vector_type(16))) _Float16 v16h;
typedef __attribute__((ext_vector_type(8)))  _Float16 v8h;
typedef __attribute__((ext_vector_type(8)))  float    v8f;

#define DEV static __device__ __forceinline__

// ---------------------------------------------------------------- WMMA helpers

DEV v8f wmma16(v16h a, v16h b, v8f c) {
  // D = A(16x32 f16) x B(32x16 f16) + C(16x16 f32)
  return __builtin_amdgcn_wmma_f32_16x16x32_f16(false, a, false, b, (short)0, c,
                                                false, false);
}

DEV v16h make_v16(v8h lo, v8h hi) {
  v16h o;
#pragma unroll
  for (int e = 0; e < 8; ++e) { o[e] = lo[e]; o[e + 8] = hi[e]; }
  return o;
}

// A fragment 16x32: element e<8 -> k = 8*(lane/16)+e ; e>=8 -> k = 16+8*(lane/16)+(e-8)
// base points at (row0, k0) of a row-major [*, ld] f16 buffer (ld multiple of 8).
DEV v16h load_a(const half_t* base, int ld) {
  int lane = threadIdx.x & 31;
  const half_t* p = base + (size_t)(lane & 15) * ld + ((lane >> 4) << 3);
  return make_v16(*(const v8h*)p, *(const v8h*)(p + 16));
}

// B fragment 32x16 where B[k][n] = base[n*ld + k] (n-major storage, contiguous k).
DEV v16h load_b(const half_t* base, int ld) {
  int lane = threadIdx.x & 31;
  const half_t* p = base + (size_t)(lane & 15) * ld + ((lane >> 4) << 3);
  return make_v16(*(const v8h*)p, *(const v8h*)(p + 16));
}

// B fragment from Er (f16 [1024,64]) rows rb..rb+15 (clamped), k-half k0.
DEV v16h load_b_er(const half_t* er16, int rb, int k0) {
  int lane = threadIdx.x & 31;
  int row = rb + (lane & 15);
  row = row < 0 ? 0 : (row > 1023 ? 1023 : row);
  const half_t* p = er16 + (size_t)row * 64 + k0 + ((lane >> 4) << 3);
  return make_v16(*(const v8h*)p, *(const v8h*)(p + 16));
}

// ---------------------------------------------------------------- small kernels

// Er f32 -> f16 copy (4*1024*64 elements)
__global__ __launch_bounds__(256) void k_cvt_er(const float* __restrict__ er,
                                                half_t* __restrict__ er16) {
  int i = blockIdx.x * 256 + threadIdx.x;
  er16[i] = (half_t)er[i];
}

// W [K,N] f32 -> Wt [Nalloc, Kp] f16 (transposed, zero padded). 32x32 LDS tiles.
__global__ __launch_bounds__(256) void k_cvt_wt(const float* __restrict__ W,
                                                half_t* __restrict__ Wt,
                                                int K, int N, int Kp) {
  __shared__ float tile[32][33];
  int kb = blockIdx.x * 32, nb = blockIdx.y * 32;
  int tx = threadIdx.x & 31, ty = threadIdx.x >> 5;
#pragma unroll
  for (int i = 0; i < 4; ++i) {
    int k = kb + ty + i * 8, n = nb + tx;
    tile[ty + i * 8][tx] = (k < K && n < N) ? W[(size_t)k * N + n] : 0.f;
  }
  __syncthreads();
#pragma unroll
  for (int i = 0; i < 4; ++i) {
    int n = nb + ty + i * 8, k = kb + tx;
    Wt[(size_t)n * Kp + k] = (half_t)tile[tx][ty + i * 8];
  }
}

// embeddings: gather + concat -> xpad f16 [2048, 288] (4 zero pad cols)
__global__ __launch_bounds__(256) void k_embed(
    const int* __restrict__ st, const int* __restrict__ ec,
    const int* __restrict__ oc, const int* __restrict__ ea,
    const int* __restrict__ oa, const float* __restrict__ gs,
    const float* __restrict__ ct, const float* __restrict__ stab,
    const float* __restrict__ ctab, const float* __restrict__ atab,
    half_t* __restrict__ xpad) {
  int t = blockIdx.x;
  int si = st[t], eci = ec[t], oci = oc[t], eai = ea[t], oai = oa[t];
  for (int c = threadIdx.x; c < 288; c += 256) {
    float v;
    if      (c < 4)   v = stab[si * 4 + c];
    else if (c < 16)  v = ctab[eci * 12 + c - 4];
    else if (c < 28)  v = ctab[oci * 12 + c - 16];
    else if (c < 60)  v = atab[eai * 32 + c - 28];
    else if (c < 92)  v = atab[oai * 32 + c - 60];
    else if (c < 220) v = gs[(size_t)t * 128 + c - 92];
    else if (c < 284) v = ct[(size_t)t * 64 + c - 220];
    else              v = 0.f;
    xpad[(size_t)t * 288 + c] = (half_t)v;
  }
}

// LayerNorm over C of f32 [2048, ldin]; writes f16 [*, ldo] zero-padded to Cp
// and/or f32 [*, ldo].
__global__ __launch_bounds__(256) void k_ln(const float* __restrict__ in, int ldin,
                                            const float* __restrict__ g,
                                            const float* __restrict__ b,
                                            half_t* __restrict__ o16,
                                            float* __restrict__ o32, int ldo,
                                            int C, int Cp) {
  __shared__ float red[8];
  int row = blockIdx.x;
  const float* xr = in + (size_t)row * ldin;
  int lane = threadIdx.x & 31, wid = threadIdx.x >> 5;
  float s = 0.f;
  for (int c = threadIdx.x; c < C; c += 256) s += xr[c];
  s += __shfl_xor(s, 1);  s += __shfl_xor(s, 2);  s += __shfl_xor(s, 4);
  s += __shfl_xor(s, 8);  s += __shfl_xor(s, 16);
  if (lane == 0) red[wid] = s;
  __syncthreads();
  float tot = 0.f;
#pragma unroll
  for (int i = 0; i < 8; ++i) tot += red[i];
  float mean = tot / (float)C;
  __syncthreads();
  float s2 = 0.f;
  for (int c = threadIdx.x; c < C; c += 256) { float d = xr[c] - mean; s2 += d * d; }
  s2 += __shfl_xor(s2, 1); s2 += __shfl_xor(s2, 2); s2 += __shfl_xor(s2, 4);
  s2 += __shfl_xor(s2, 8); s2 += __shfl_xor(s2, 16);
  if (lane == 0) red[wid] = s2;
  __syncthreads();
  float tv = 0.f;
#pragma unroll
  for (int i = 0; i < 8; ++i) tv += red[i];
  float rstd = rsqrtf(tv / (float)C + 1e-5f);
  for (int c = threadIdx.x; c < Cp; c += 256) {
    float v = 0.f;
    if (c < C) v = (xr[c] - mean) * rstd * g[c] + b[c];
    if (o16) o16[(size_t)row * ldo + c] = (half_t)v;
    if (o32 && c < C) o32[(size_t)row * ldo + c] = v;
  }
}

// transpose V slice of qkv16 -> vT [B][H][64][1024] f16
__global__ __launch_bounds__(256) void k_vtrans(const half_t* __restrict__ qkv,
                                                half_t* __restrict__ vT) {
  __shared__ __attribute__((aligned(16))) half_t tile[32][33];
  int bh = blockIdx.z, b = bh >> 4, h = bh & 15;
  int lb = blockIdx.x * 32, cb = blockIdx.y * 32;
  int tx = threadIdx.x & 31, ty = threadIdx.x >> 5;
#pragma unroll
  for (int i = 0; i < 4; ++i) {
    int l = lb + ty + i * 8, c = cb + tx;
    tile[ty + i * 8][tx] = qkv[(size_t)(b * 1024 + l) * 3072 + 2048 + h * 64 + c];
  }
  __syncthreads();
#pragma unroll
  for (int i = 0; i < 4; ++i) {
    int c = cb + ty + i * 8, l = lb + tx;
    vT[(size_t)((b * 16 + h) * 64 + c) * 1024 + l] = tile[tx][ty + i * 8];
  }
}

// ---------------------------------------------------------------- WMMA GEMM
// out = epilogue(A[2048,Kp] @ Wt^T + bias).  Wt stored [Nalloc, Kp] (n-major).
// Block = 256 thr = 8 waves (4x2), tile 128x64, wave tile 32x32.
__global__ __launch_bounds__(256) void k_gemm(
    const half_t* __restrict__ A, const half_t* __restrict__ Wt,
    const float* __restrict__ bias, const float* __restrict__ res, int ldr,
    float* __restrict__ o32a, int ld32a, float* __restrict__ o32b, int ld32b,
    half_t* __restrict__ o16, int ld16, int n16, int N, int Kp, int gelu) {
  int wid = threadIdx.x >> 5, lane = threadIdx.x & 31;
  int wm = wid & 3, wn = wid >> 2;
  int m0 = blockIdx.x * 128 + wm * 32;
  int n0 = blockIdx.y * 64 + wn * 32;
  const v8f vzero = {};
  v8f acc[2][2];
#pragma unroll
  for (int i = 0; i < 2; ++i)
#pragma unroll
    for (int j = 0; j < 2; ++j) acc[i][j] = vzero;
  const half_t* Ar0 = A + (size_t)m0 * Kp;
  const half_t* Ar1 = A + (size_t)(m0 + 16) * Kp;
  const half_t* Wr0 = Wt + (size_t)n0 * Kp;
  const half_t* Wr1 = Wt + (size_t)(n0 + 16) * Kp;
  for (int kk = 0; kk < Kp; kk += 32) {
    v16h a0 = load_a(Ar0 + kk, Kp);
    v16h a1 = load_a(Ar1 + kk, Kp);
    v16h b0 = load_b(Wr0 + kk, Kp);
    v16h b1 = load_b(Wr1 + kk, Kp);
    acc[0][0] = wmma16(a0, b0, acc[0][0]);
    acc[0][1] = wmma16(a0, b1, acc[0][1]);
    acc[1][0] = wmma16(a1, b0, acc[1][0]);
    acc[1][1] = wmma16(a1, b1, acc[1][1]);
  }
  int nl = lane & 15, hf = lane >> 4;
#pragma unroll
  for (int i = 0; i < 2; ++i)
#pragma unroll
    for (int j = 0; j < 2; ++j)
#pragma unroll
      for (int r = 0; r < 8; ++r) {
        int row = m0 + i * 16 + r + hf * 8;
        int col = n0 + j * 16 + nl;
        float v = acc[i][j][r];
        if (col < N) {
          if (bias) v += bias[col];
          if (gelu) v = 0.5f * v * (1.f + erff(v * 0.70710678f));
          if (res)  v += res[(size_t)row * ldr + col];
          if (o32a) o32a[(size_t)row * ld32a + col] = v;
          if (o32b) o32b[(size_t)row * ld32b + col] = v;
          if (o16)  o16[(size_t)row * ld16 + col] = (half_t)v;
        } else if (o16 && col < n16) {
          o16[(size_t)row * ld16 + col] = (half_t)0.f;   // zero K-pad for next GEMM
        }
      }
}

// ---------------------------------------------------------------- attention
// Flash attention + Music-Transformer relative term, all in WMMA.
// grid (L/64, H, B), block 128 (4 waves), each wave owns 16 query rows.
__global__ __launch_bounds__(128) void k_attn(const half_t* __restrict__ qkv,
                                              const half_t* __restrict__ vT,
                                              const half_t* __restrict__ er16,
                                              half_t* __restrict__ y) {
  __shared__ __attribute__((aligned(16))) float  relS[4][16][32];
  __shared__ __attribute__((aligned(16))) half_t pS[4][16][32];
  int wid = threadIdx.x >> 5, lane = threadIdx.x & 31;
  int nl = lane & 15, hf = lane >> 4;
  int b = blockIdx.z, h = blockIdx.y;
  int i0 = blockIdx.x * 64 + wid * 16;
  const half_t* Qb = qkv + (size_t)(b * 1024 + i0) * 3072 + h * 64;
  const half_t* Kb = qkv + (size_t)(b * 1024) * 3072 + 1024 + h * 64;
  const half_t* Vt = vT + (size_t)((b * 16 + h) * 64) * 1024;
  v16h qa0 = load_a(Qb, 3072);
  v16h qa1 = load_a(Qb + 32, 3072);
  const v8f vzero = {};
  v8f o[4];
#pragma unroll
  for (int t = 0; t < 4; ++t) o[t] = vzero;
  float mrow[8], lrow[8];
#pragma unroll
  for (int r = 0; r < 8; ++r) { mrow[r] = -3.0e38f; lrow[r] = 0.f; }
  int nsteps = (i0 + 47) >> 5;
  for (int js = 0; js < nsteps; ++js) {
    int j0 = js * 32;
#pragma unroll
    for (int sub = 0; sub < 2; ++sub) {
      int jsub = j0 + (sub << 4);
      v8f s = vzero;
      s = wmma16(qa0, load_b(Kb + (size_t)jsub * 3072, 3072), s);
      s = wmma16(qa1, load_b(Kb + (size_t)jsub * 3072 + 32, 3072), s);
      // Srel[i,j] = QEr[i, 1023-i+j]; rb chosen so needed cols live in [0,31)
      int rb = 1008 - i0 + jsub;
      v8f u = vzero, w = vzero;
      u = wmma16(qa0, load_b_er(er16, rb, 0), u);
      u = wmma16(qa1, load_b_er(er16, rb, 32), u);
      w = wmma16(qa0, load_b_er(er16, rb + 16, 0), w);
      w = wmma16(qa1, load_b_er(er16, rb + 16, 32), w);
#pragma unroll
      for (int r = 0; r < 8; ++r) {
        relS[wid][r + hf * 8][nl]      = u[r];
        relS[wid][r + hf * 8][nl + 16] = w[r];
      }
      float p[8];
#pragma unroll
      for (int r = 0; r < 8; ++r) {
        int ro = r + hf * 8;
        float sv = (s[r] + relS[wid][ro][nl - ro + 15]) * 0.125f;
        if (jsub + nl > i0 + ro) sv = -3.0e38f;   // causal mask
        p[r] = sv;
      }
      // online softmax: row stats across the 16 lanes of each half
#pragma unroll
      for (int r = 0; r < 8; ++r) {
        float mv = p[r];
        mv = fmaxf(mv, __shfl_xor(mv, 1));
        mv = fmaxf(mv, __shfl_xor(mv, 2));
        mv = fmaxf(mv, __shfl_xor(mv, 4));
        mv = fmaxf(mv, __shfl_xor(mv, 8));
        float mnew  = fmaxf(mrow[r], mv);
        float alpha = __expf(mrow[r] - mnew);
        float pe    = __expf(p[r] - mnew);
        float ps = pe;
        ps += __shfl_xor(ps, 1); ps += __shfl_xor(ps, 2);
        ps += __shfl_xor(ps, 4); ps += __shfl_xor(ps, 8);
        lrow[r] = lrow[r] * alpha + ps;
        mrow[r] = mnew;
#pragma unroll
        for (int t = 0; t < 4; ++t) o[t][r] *= alpha;
        pS[wid][r + hf * 8][(sub << 4) + nl] = (half_t)pe;
      }
    }
    // O += P(16x32) @ V(32x64)   (P staged via LDS into A-fragment layout)
    v16h pa = load_a(&pS[wid][0][0], 32);
#pragma unroll
    for (int t = 0; t < 4; ++t)
      o[t] = wmma16(pa, load_b(Vt + (size_t)(t * 16) * 1024 + j0, 1024), o[t]);
  }
  half_t* yb = y + (size_t)(b * 1024 + i0) * 1024 + h * 64;
#pragma unroll
  for (int t = 0; t < 4; ++t)
#pragma unroll
    for (int r = 0; r < 8; ++r)
      yb[(size_t)(r + hf * 8) * 1024 + t * 16 + nl] = (half_t)(o[t][r] / lrow[r]);
}

// ---------------------------------------------------------------- host driver

static inline int ru(int x, int m) { return (x + m - 1) / m * m; }

extern "C" void kernel_launch(void* const* d_in, const int* in_sizes, int n_in,
                              void* d_out, int out_size, void* d_ws, size_t ws_size,
                              hipStream_t stream) {
  (void)in_sizes; (void)n_in; (void)out_size; (void)ws_size;
  // ---- inputs (setup_inputs order)
  const int*   stage = (const int*)d_in[0];
  const int*   egoc  = (const int*)d_in[1];
  const int*   oppc  = (const int*)d_in[2];
  const int*   egoa  = (const int*)d_in[3];
  const int*   oppa  = (const int*)d_in[4];
  const float* gs    = (const float*)d_in[5];
  const float* ctl   = (const float*)d_in[6];
  const float* stab  = (const float*)d_in[7];
  const float* ctab  = (const float*)d_in[8];
  const float* atab  = (const float*)d_in[9];
  const float* proj_w = (const float*)d_in[10];
  const float* proj_b = (const float*)d_in[11];
  const float* ln1_g = (const float*)d_in[12];
  const float* ln1_b = (const float*)d_in[13];
  const float* attn_w = (const float*)d_in[14];
  const float* attn_b = (const float*)d_in[15];
  const float* atp_w = (const float*)d_in[16];
  const float* atp_b = (const float*)d_in[17];
  const float* Er    = (const float*)d_in[18];
  const float* ln2_g = (const float*)d_in[19];
  const float* ln2_b = (const float*)d_in[20];
  const float* fc_w  = (const float*)d_in[21];
  const float* fc_b  = (const float*)d_in[22];
  const float* fcp_w = (const float*)d_in[23];
  const float* fcp_b = (const float*)d_in[24];
  const float* lnf_g = (const float*)d_in[25];
  const float* lnf_b = (const float*)d_in[26];
  float* out = (float*)d_out;

  // ---- workspace carve-up
  char* wsp = (char*)d_ws;
  size_t off = 0;
  auto carve = [&](size_t bytes) {
    void* p = wsp + off; off += (bytes + 255) & ~(size_t)255; return p;
  };
  float*  x     = (float*) carve(2048ull * 1024 * 4);
  float*  hx    = (float*) carve(2048ull * 1088 * 4);
  half_t* ln16  = (half_t*)carve(2048ull * 1088 * 2);
  half_t* act16 = (half_t*)carve(2048ull * 4096 * 2);
  half_t* qkv16 = (half_t*)carve(2048ull * 3072 * 2);
  half_t* vT    = (half_t*)carve(2048ull * 1024 * 2);
  half_t* y16   = (half_t*)carve(2048ull * 1024 * 2);
  half_t* xpad  = (half_t*)carve(2048ull * 288 * 2);
  half_t* er16  = (half_t*)carve(4ull * 1024 * 64 * 2);
  half_t* wt    = (half_t*)carve(4096ull * 1024 * 2);

  auto ln = [&](const float* in, int ldin, const float* g, const float* b,
                half_t* o16, float* o32, int ldo, int C, int Cp) {
    k_ln<<<2048, 256, 0, stream>>>(in, ldin, g, b, o16, o32, ldo, C, Cp);
  };
  // convert weight -> wt, then GEMM (A's leading dim == Kp by construction)
  auto mm = [&](const float* Wf32, const half_t* A, int Kp, int K, int N,
                const float* bias, const float* res, int ldr,
                float* a32, int lda32, float* b32, int ldb32,
                half_t* o16, int ld16, int n16, int gelu) {
    int ncov = (n16 > N ? n16 : N);
    int Nalloc = ru(ncov, 64);
    k_cvt_wt<<<dim3(Kp / 32, Nalloc / 32), 256, 0, stream>>>(Wf32, wt, K, N, Kp);
    k_gemm<<<dim3(16, Nalloc / 64), 256, 0, stream>>>(
        A, wt, bias, res, ldr, a32, lda32, b32, ldb32, o16, ld16, n16, N, Kp, gelu);
  };

  // ---- preamble
  k_cvt_er<<<(4 * 1024 * 64) / 256, 256, 0, stream>>>(Er, er16);
  k_embed<<<2048, 256, 0, stream>>>(stage, egoc, oppc, egoa, oppa, gs, ctl,
                                    stab, ctab, atab, xpad);
  // input projection: x = xcat @ proj_w + proj_b
  mm(proj_w, xpad, 288, 284, 1024, proj_b, nullptr, 0,
     x, 1024, nullptr, 0, nullptr, 0, 0, 0);

  // ---- transformer layers
  for (int i = 0; i < 4; ++i) {
    ln(x, 1024, ln1_g + i * 1024, ln1_b + i * 1024, ln16, nullptr, 1024, 1024, 1024);
    mm(attn_w + (size_t)i * 1024 * 3072, ln16, 1024, 1024, 3072,
       attn_b + i * 3072, nullptr, 0, nullptr, 0, nullptr, 0, qkv16, 3072, 3072, 0);
    k_vtrans<<<dim3(32, 2, 32), 256, 0, stream>>>(qkv16, vT);
    k_attn<<<dim3(16, 16, 2), 128, 0, stream>>>(qkv16, vT, er16 + i * 1024 * 64, y16);
    mm(atp_w + (size_t)i * 1024 * 1024, y16, 1024, 1024, 1024,
       atp_b + i * 1024, x, 1024, x, 1024, nullptr, 0, nullptr, 0, 0, 0);
    ln(x, 1024, ln2_g + i * 1024, ln2_b + i * 1024, ln16, nullptr, 1024, 1024, 1024);
    mm(fc_w + (size_t)i * 1024 * 4096, ln16, 1024, 1024, 4096,
       fc_b + i * 4096, nullptr, 0, nullptr, 0, nullptr, 0, act16, 4096, 4096, 1);
    mm(fcp_w + (size_t)i * 4096 * 1024, act16, 4096, 4096, 1024,
       fcp_b + i * 1024, x, 1024, x, 1024, nullptr, 0, nullptr, 0, 0, 0);
  }

  // ---- final LN into hx[:, 0:1024]
  ln(x, 1024, lnf_g, lnf_b, nullptr, hx, 1088, 1024, 1024);

  // ---- cascaded heads: hx cols [0:1024)=x, [1024:1045)=cs, [1045:1066)=ms, [1066:1074)=bt
  int cins[4]   = {1024, 1045, 1066, 1074};
  int couts[4]  = {21, 21, 8, 5};
  int outoffs[4] = {0, 21, 42, 50};
  int base = 27;
  for (int k = 0; k < 4; ++k) {
    int cin = cins[k], mid = cin / 2, cout = couts[k];
    int Kp1 = ru(cin, 32), Np1 = ru(mid, 32);
    const float* hg = (const float*)d_in[base + 0];
    const float* hb = (const float*)d_in[base + 1];
    const float* w1 = (const float*)d_in[base + 2];
    const float* b1 = (const float*)d_in[base + 3];
    const float* w2 = (const float*)d_in[base + 4];
    const float* b2 = (const float*)d_in[base + 5];
    base += 6;
    ln(hx, 1088, hg, hb, ln16, nullptr, Kp1, cin, Kp1);
    mm(w1, ln16, Kp1, cin, mid, b1, nullptr, 0,
       nullptr, 0, nullptr, 0, act16, Np1, Np1, 1);
    mm(w2, act16, Np1, mid, cout, b2, nullptr, 0,
       hx + cins[k], 1088, out + outoffs[k], 55, nullptr, 0, 0, 0);
  }
}
